// CMRET_79594333929935
// MI455X (gfx1250) — compile-verified, matrix operand order checked by reading, not run on hardware
//
#include <hip/hip_runtime.h>
#include <hip/hip_bf16.h>
#include <cstdint>

// ---------------- model constants ----------------
#define NA   4096
#define FD   128
#define HH   4
#define KN   15
#define KP   16      // padded K for WMMA tiles
#define NKR  50
#define NKP  64      // padded RBF dim for WMMA (k-dim)
#define LL   6
#define DH   32
#define NQKV 640     // q(128) + k(128) + v(384)
#define NEKV 512     // ek(128) + ev(384)
#define SCALE 0.125f // 1/sqrt(DH*TEMP) = 1/sqrt(64)

typedef __attribute__((ext_vector_type(16))) _Float16 v16h;
typedef __attribute__((ext_vector_type(8)))  float    v8f;

// ---- async global->LDS staging (CDNA5 GLOBAL_LOAD_ASYNC_TO_LDS_B128) ----
#if __has_builtin(__builtin_amdgcn_global_load_async_to_lds_b128)
#define HAVE_ASYNC_LDS 1
typedef int v4i_vs __attribute__((vector_size(16)));
typedef __attribute__((address_space(1))) v4i_vs as1_v4i;
typedef __attribute__((address_space(3))) v4i_vs as3_v4i;
__device__ __forceinline__ void stage16B(void* dstLds, const void* srcGlobal) {
  __builtin_amdgcn_global_load_async_to_lds_b128(
      (as1_v4i*)srcGlobal, (as3_v4i*)dstLds, 0, 0);
}
#else
#define HAVE_ASYNC_LDS 0
__device__ __forceinline__ void stage16B(void* dstLds, const void* srcGlobal) {
  *(float4*)dstLds = *(const float4*)srcGlobal;
}
#endif

__device__ __forceinline__ void stage_wait() {
#if HAVE_ASYNC_LDS
#if __has_builtin(__builtin_amdgcn_s_wait_asynccnt)
  __builtin_amdgcn_s_wait_asynccnt(0);
#elif defined(__gfx1250__)
  asm volatile("s_wait_asynccnt 0" ::: "memory");
#endif
#endif
}

// ---- WMMA fragment helpers (CDNA5 ISA 7.12.2 layouts, wave32) ----
// 16-bit A (16xK) / B (Kx16): lane L -> m/n = L&15, hi = L>=16.
// element e<8 : k = e + hi*8 ; element e>=8 : k = e + 8 + hi*8
__device__ __forceinline__ int frag_k_map(int e, int hi) {
  return (e < 8 ? e : e + 8) + (hi ? 8 : 0);
}

// A fragment from an LDS tile (row-major halfs, row stride `stride`)
__device__ __forceinline__ v16h load_A_lds(const _Float16* base, int stride, int kOff) {
  int lane = threadIdx.x & 31;
  int m = lane & 15, hi = (lane >> 4) & 1;
  const _Float16* row = base + m * stride + kOff + hi * 8;
  union { v16h v; float4 q[2]; } u;
  u.q[0] = *(const float4*)(row);        // halves k .. k+7   -> two ds_load_b128
  u.q[1] = *(const float4*)(row + 16);   // halves k+16..k+23
  return u.v;
}

// B fragment from a pre-swizzled weight image: per (ktile,ntile,lane) the 16
// halves are contiguous -> a single 32B global load.
__device__ __forceinline__ v16h load_B_sw(const _Float16* W, int ntiles, int kt, int nt) {
  int lane = threadIdx.x & 31;
  return *(const v16h*)(W + (((size_t)kt * ntiles + nt) * 32 + lane) * 16);
}

// ---------------- prep kernels ----------------

// Pack Wq|Wk|Wv (f32) into a pre-swizzled f16 image:  L x (4 ktiles x 40 ntiles x 32 lanes x 16 elems)
__global__ void k_prep_wqkv(const float* __restrict__ Wq, const float* __restrict__ Wk,
                            const float* __restrict__ Wv, _Float16* __restrict__ Wh) {
  int g = blockIdx.x * 256 + threadIdx.x;
  const int per_l = FD * NQKV; // 81920
  if (g >= LL * per_l) return;
  int l = g / per_l, g2 = g % per_l;
  int e = g2 & 15, lane = (g2 >> 4) & 31, rem = g2 >> 9;
  int nt = rem % (NQKV / 16), kt = rem / (NQKV / 16);
  int k = kt * 32 + frag_k_map(e, lane >> 4);
  int n = nt * 16 + (lane & 15);
  float v;
  if (n < 128)       v = Wq[((size_t)l * FD + k) * FD + n];
  else if (n < 256)  v = Wk[((size_t)l * FD + k) * FD + (n - 128)];
  else               v = Wv[((size_t)l * FD + k) * 384 + (n - 256)];
  Wh[g] = (_Float16)v;
}

// Pack We_k|We_v (f32, k-dim 50 padded to 64) into swizzled f16 image: L x (2 kt x 32 nt x 32 x 16)
__global__ void k_prep_we(const float* __restrict__ Wek, const float* __restrict__ Wev,
                          _Float16* __restrict__ Wh) {
  int g = blockIdx.x * 256 + threadIdx.x;
  const int per_l = NKP * NEKV; // 32768
  if (g >= LL * per_l) return;
  int l = g / per_l, g2 = g % per_l;
  int e = g2 & 15, lane = (g2 >> 4) & 31, rem = g2 >> 9;
  int nt = rem % (NEKV / 16), kt = rem / (NEKV / 16);
  int k = kt * 32 + frag_k_map(e, lane >> 4);
  int n = nt * 16 + (lane & 15);
  float v = 0.f;
  if (k < NKR) v = (n < 128) ? Wek[((size_t)l * NKR + k) * FD + n]
                             : Wev[((size_t)l * NKR + k) * 384 + (n - 128)];
  Wh[g] = (_Float16)v;
}

// s = emb[Z]; o = 0; v = 0; out = 0
__global__ void k_init_state(const int* __restrict__ Z, const float* __restrict__ emb,
                             float* __restrict__ s, _Float16* __restrict__ sh,
                             float* __restrict__ o, float* __restrict__ va,
                             float* __restrict__ out) {
  int g = blockIdx.x * 256 + threadIdx.x;
  if (g < NA * FD) {
    int i = g >> 7, f = g & 127;
    float v = emb[(size_t)Z[i] * FD + f];
    s[g] = v; sh[g] = (_Float16)v; o[g] = 0.f;
  }
  if (g < NA * 3 * FD) va[g] = 0.f;
  if (g == 0) out[0] = 0.f;
}

// ---------------- top-K neighbor search (one block / atom) ----------------
__global__ void k_topk(const float* __restrict__ R, const float* __restrict__ batch,
                       int* __restrict__ idxs, float* __restrict__ cutb,
                       float* __restrict__ dbuf, float* __restrict__ vecn) {
  __shared__ float d2s[NA];
  __shared__ float rmin[256];
  __shared__ int   ridx[256];
  __shared__ int   selS[KP];
  int i = blockIdx.x, tid = threadIdx.x;
  float xi = R[i * 3], yi = R[i * 3 + 1], zi = R[i * 3 + 2];
  float bi = batch[i];
  for (int c = tid; c < NA; c += 256) {
    float dx = R[c * 3] - xi, dy = R[c * 3 + 1] - yi, dz = R[c * 3 + 2] - zi;
    float d2 = dx * dx + dy * dy + dz * dz;
    bool msk = (c == i) || (bi * batch[c] == 0.f);
    d2s[c] = msk ? 1e9f : d2;
  }
  __syncthreads();
  for (int it = 0; it < KN; it++) {
    float m = 1e30f; int mi = NA;
    for (int c = tid; c < NA; c += 256) {
      float v = d2s[c];
      if (v < m || (v == m && c < mi)) { m = v; mi = c; }
    }
    rmin[tid] = m; ridx[tid] = mi;
    __syncthreads();
    for (int s = 128; s; s >>= 1) {
      if (tid < s) {
        float a = rmin[tid], b = rmin[tid + s];
        int ia = ridx[tid], ib = ridx[tid + s];
        if (b < a || (b == a && ib < ia)) { rmin[tid] = b; ridx[tid] = ib; }
      }
      __syncthreads();
    }
    if (tid == 0) { int sel = ridx[0]; selS[it] = sel; d2s[sel] = 1e9f; }
    __syncthreads();
  }
  if (tid < KP) {
    if (tid < KN) {
      int j = selS[tid];
      float dx = R[j * 3] - xi, dy = R[j * 3 + 1] - yi, dz = R[j * 3 + 2] - zi;
      float d = sqrtf(dx * dx + dy * dy + dz * dz + 1e-12f);
      float cut = (d <= 5.0f) ? 0.5f * (__cosf(3.14159265358979f * d * 0.2f) + 1.0f) : 0.f;
      idxs[i * KP + tid] = j;
      dbuf[i * KP + tid] = d;
      cutb[i * KP + tid] = cut;
      vecn[(i * KP + tid) * 3 + 0] = dx / d;
      vecn[(i * KP + tid) * 3 + 1] = dy / d;
      vecn[(i * KP + tid) * 3 + 2] = dz / d;
    } else { // padded edge: dead (cut==0), WMMA rows for it are zero
      idxs[i * KP + tid] = 0; dbuf[i * KP + tid] = 0.f; cutb[i * KP + tid] = 0.f;
      vecn[(i * KP + tid) * 3 + 0] = 0.f;
      vecn[(i * KP + tid) * 3 + 1] = 0.f;
      vecn[(i * KP + tid) * 3 + 2] = 0.f;
    }
  }
}

// ---------------- RBF edge features (padded 16x64 f16 tiles) ----------------
__global__ void k_rbf(const float* __restrict__ dbuf, const float* __restrict__ cutb,
                      _Float16* __restrict__ eh) {
  int g = blockIdx.x * 256 + threadIdx.x;   // NA*16*64
  int c = g & 63, ek = (g >> 6) & 15, a = g >> 10;
  if (a >= NA) return;
  float v = 0.f;
  if (ek < KN && c < NKR) {
    float d = dbuf[a * KP + ek], cut = cutb[a * KP + ek];
    float mu = (float)c * (5.0f / 49.0f);   // linspace(0, CUTOFF, 50)
    float t = d - mu;
    v = __expf(-t * t * (1.0f / (2.0f * 0.1f * 0.1f))) * cut; // delta = 0.1
  }
  eh[g] = (_Float16)v;
}

// ---------------- qkv GEMM: (4096x128) x (128x640), f16 WMMA ----------------
// grid.x = NA/16 row tiles, 256 threads = 8 waves, each wave does 5 n-tiles.
// A tile staged to LDS with async global->LDS DMA; B fragments stream from L2.
__global__ void k_gemm_qkv(const _Float16* __restrict__ sh,
                           const _Float16* __restrict__ Wl,   // swizzled, 4x40 tiles
                           float* __restrict__ qkv) {
  __shared__ __align__(16) _Float16 At[16 * FD];
  int rowT = blockIdx.x, tid = threadIdx.x;
  { // cooperative 16x128 f16 tile stage (16B per thread, async to LDS)
    int r = tid >> 4, c8 = (tid & 15) * 8;
    stage16B(&At[r * FD + c8], &sh[((size_t)rowT * 16 + r) * FD + c8]);
  }
  stage_wait();
  __syncthreads();
  int w = tid >> 5, lane = tid & 31, hi = (lane >> 4) & 1;
  for (int t = 0; t < 5; t++) {
    int nt = w + t * 8;                           // 0..39
    __builtin_prefetch(Wl + ((size_t)nt * 32) * 16, 0, 1);
    v8f acc = {};
#pragma unroll
    for (int ks = 0; ks < 4; ks++) {
      v16h a = load_A_lds(At, FD, ks * 32);
      v16h b = load_B_sw(Wl, NQKV / 16, ks, nt);
      acc = __builtin_amdgcn_wmma_f32_16x16x32_f16(false, a, false, b,
                                                   (short)0, acc, false, false);
    }
    int n = nt * 16 + (lane & 15);
#pragma unroll
    for (int r = 0; r < 8; r++) {
      int m = rowT * 16 + r + hi * 8;
      qkv[(size_t)m * NQKV + n] = acc[r];
    }
  }
}

// ------- fused edge GEMM (WMMA) + silu + K=15 attention + updates -------
// one block (128 threads = 4 waves) per atom; wave == head (DH == 32).
__global__ void k_attn(const float* __restrict__ qkv, const _Float16* __restrict__ eh,
                       const _Float16* __restrict__ Wel,  // swizzled, 2x32 tiles
                       const float* __restrict__ bek, const float* __restrict__ bev,
                       const int* __restrict__ idxs, const float* __restrict__ cutb,
                       const float* __restrict__ vecn,
                       const float* __restrict__ vin, float* __restrict__ vout,
                       float* __restrict__ s, _Float16* __restrict__ sh,
                       float* __restrict__ o) {
  __shared__ __align__(16) _Float16 Et[16 * NKP];   // 2 KB
  __shared__ float ekv[16 * NEKV];                  // 32 KB: silu(e@We + b)
  __shared__ int   idxS[KP];
  __shared__ float cutS[KP];
  __shared__ float vnS[KP * 3];
  int i = blockIdx.x, tid = threadIdx.x;
  { // stage padded 16x64 edge tile (16B per thread, async to LDS)
    int r = tid >> 3, c8 = (tid & 7) * 8;
    stage16B(&Et[r * NKP + c8], &eh[((size_t)i * KP + r) * NKP + c8]);
  }
  if (tid < KP) { idxS[tid] = idxs[i * KP + tid]; cutS[tid] = cutb[i * KP + tid]; }
  if (tid < KP * 3) vnS[tid] = vecn[i * KP * 3 + tid];
  stage_wait();
  __syncthreads();
  // ---- WMMA: ekv[16x512] = silu(e(16x64) @ We(64x512) + bias) ----
  int w = tid >> 5, lane = tid & 31, hi = (lane >> 4) & 1;
  for (int t = 0; t < 8; t++) {
    int nt = w * 8 + t;                               // 0..31
    __builtin_prefetch(Wel + ((size_t)nt * 32) * 16, 0, 1);
    v8f acc = {};
#pragma unroll
    for (int ks = 0; ks < 2; ks++) {
      v16h a = load_A_lds(Et, NKP, ks * 32);
      v16h b = load_B_sw(Wel, NEKV / 16, ks, nt);
      acc = __builtin_amdgcn_wmma_f32_16x16x32_f16(false, a, false, b,
                                                   (short)0, acc, false, false);
    }
    int n = nt * 16 + (lane & 15);
    float bias = (n < 128) ? bek[n] : bev[n - 128];
#pragma unroll
    for (int r = 0; r < 8; r++) {
      float x = acc[r] + bias;
      ekv[(r + hi * 8) * NEKV + n] = x / (1.f + __expf(-x));   // silu
    }
  }
  __syncthreads();
  // ---- attention (VALU), channel f = tid; head = tid/32 = wave ----
  int f = tid;
  float q = qkv[(size_t)i * NQKV + f];
  float logit[KN];
  float mx = -1e30f;
#pragma unroll
  for (int k = 0; k < KN; k++) {
    int j = idxS[k];
    float kh = qkv[(size_t)j * NQKV + 128 + f] * ekv[k * NEKV + f];
    float p = q * kh;
#pragma unroll
    for (int off = 16; off; off >>= 1) p += __shfl_xor(p, off, 32);
    logit[k] = p * SCALE;
    mx = fmaxf(mx, logit[k]);
  }
  float den = 0.f, wgt[KN];
#pragma unroll
  for (int k = 0; k < KN; k++) { wgt[k] = __expf(logit[k] - mx); den += wgt[k]; }
  float inv = 1.f / den;
  float ds = 0.f, dv0 = 0.f, dv1 = 0.f, dv2 = 0.f;
#pragma unroll
  for (int k = 0; k < KN; k++) {
    int j = idxS[k];
    float a = wgt[k] * inv * cutS[k];
    float m1 = qkv[(size_t)j * NQKV + 256 + f] * ekv[k * NEKV + 128 + f];
    float m2 = qkv[(size_t)j * NQKV + 384 + f] * ekv[k * NEKV + 256 + f];
    float m3 = qkv[(size_t)j * NQKV + 512 + f] * ekv[k * NEKV + 384 + f];
    ds += a * m1;
    float wm2 = a * m2, wm3 = a * m3;
    dv0 += wm2 * vnS[k * 3 + 0] + wm3 * vin[((size_t)j * 3 + 0) * FD + f];
    dv1 += wm2 * vnS[k * 3 + 1] + wm3 * vin[((size_t)j * 3 + 1) * FD + f];
    dv2 += wm2 * vnS[k * 3 + 2] + wm3 * vin[((size_t)j * 3 + 2) * FD + f];
  }
  float sN = s[(size_t)i * FD + f] + ds;
  s[(size_t)i * FD + f] = sN;
  sh[(size_t)i * FD + f] = (_Float16)sN;
  o[(size_t)i * FD + f] += ds;
  vout[((size_t)i * 3 + 0) * FD + f] = vin[((size_t)i * 3 + 0) * FD + f] + dv0;
  vout[((size_t)i * 3 + 1) * FD + f] = vin[((size_t)i * 3 + 1) * FD + f] + dv1;
  vout[((size_t)i * 3 + 2) * FD + f] = vin[((size_t)i * 3 + 2) * FD + f] + dv2;
}

// ---------------- LayerNorm + MLP readout + energy reduction ----------------
__global__ void k_readout(const float* __restrict__ o, const float* __restrict__ lng,
                          const float* __restrict__ lnb, const float* __restrict__ Wo1,
                          const float* __restrict__ Wo2, const float* __restrict__ batch,
                          float* __restrict__ out) {
  __shared__ float on[FD];
  __shared__ float rbuf[4];
  int i = blockIdx.x, tid = threadIdx.x, w = tid >> 5;
  float x = o[(size_t)i * FD + tid];
  float p = x;
#pragma unroll
  for (int off = 16; off; off >>= 1) p += __shfl_xor(p, off, 32);
  if ((tid & 31) == 0) rbuf[w] = p;
  __syncthreads();
  float mean = (rbuf[0] + rbuf[1] + rbuf[2] + rbuf[3]) * (1.0f / FD);
  __syncthreads();
  float d = x - mean;
  p = d * d;
#pragma unroll
  for (int off = 16; off; off >>= 1) p += __shfl_xor(p, off, 32);
  if ((tid & 31) == 0) rbuf[w] = p;
  __syncthreads();
  float var = (rbuf[0] + rbuf[1] + rbuf[2] + rbuf[3]) * (1.0f / FD);
  on[tid] = d * rsqrtf(var + 1e-5f) * lng[tid] + lnb[tid];
  __syncthreads();
  float t = 0.f;
  for (int f = 0; f < FD; f++) t += on[f] * Wo1[(size_t)f * FD + tid];
  float si = t / (1.f + __expf(-t));
  p = si * Wo2[tid];
#pragma unroll
  for (int off = 16; off; off >>= 1) p += __shfl_xor(p, off, 32);
  __syncthreads();
  if ((tid & 31) == 0) rbuf[w] = p;
  __syncthreads();
  if (tid == 0) atomicAdd(out, (rbuf[0] + rbuf[1] + rbuf[2] + rbuf[3]) * batch[i]);
}

// ---------------- launch ----------------
extern "C" void kernel_launch(void* const* d_in, const int* in_sizes, int n_in,
                              void* d_out, int out_size, void* d_ws, size_t ws_size,
                              hipStream_t stream) {
  const int*   Z     = (const int*)  d_in[0];
  const float* R     = (const float*)d_in[1];
  const float* batch = (const float*)d_in[2];
  const float* emb   = (const float*)d_in[3];
  const float* Wq    = (const float*)d_in[4];
  const float* Wk    = (const float*)d_in[5];
  const float* Wv    = (const float*)d_in[6];
  const float* Wek   = (const float*)d_in[7];
  const float* bek   = (const float*)d_in[8];
  const float* Wev   = (const float*)d_in[9];
  const float* bev   = (const float*)d_in[10];
  const float* lng   = (const float*)d_in[11];
  const float* lnb   = (const float*)d_in[12];
  const float* Wo1   = (const float*)d_in[13];
  const float* Wo2   = (const float*)d_in[14];
  float* out = (float*)d_out;

  char* wp = (char*)d_ws;
  auto carve = [&](size_t bytes) -> char* {
    char* p = wp; wp += (bytes + 255) & ~(size_t)255; return p;
  };
  float*    s_     = (float*)   carve((size_t)NA * FD * 4);
  _Float16* sh_    = (_Float16*)carve((size_t)NA * FD * 2);
  float*    o_     = (float*)   carve((size_t)NA * FD * 4);
  float*    va     = (float*)   carve((size_t)NA * 3 * FD * 4);
  float*    vb     = (float*)   carve((size_t)NA * 3 * FD * 4);
  float*    qkv    = (float*)   carve((size_t)NA * NQKV * 4);
  int*      idxs   = (int*)     carve((size_t)NA * KP * 4);
  float*    cutb   = (float*)   carve((size_t)NA * KP * 4);
  float*    dbuf   = (float*)   carve((size_t)NA * KP * 4);
  float*    vecn   = (float*)   carve((size_t)NA * KP * 3 * 4);
  _Float16* eh     = (_Float16*)carve((size_t)NA * KP * NKP * 2);
  _Float16* Wqkvh  = (_Float16*)carve((size_t)LL * FD * NQKV * 2);
  _Float16* Weh    = (_Float16*)carve((size_t)LL * NKP * NEKV * 2);

  k_prep_wqkv<<<(LL * FD * NQKV + 255) / 256, 256, 0, stream>>>(Wq, Wk, Wv, Wqkvh);
  k_prep_we  <<<(LL * NKP * NEKV + 255) / 256, 256, 0, stream>>>(Wek, Wev, Weh);
  k_init_state<<<(NA * 3 * FD + 255) / 256, 256, 0, stream>>>(Z, emb, s_, sh_, o_, va, out);
  k_topk<<<NA, 256, 0, stream>>>(R, batch, idxs, cutb, dbuf, vecn);
  k_rbf <<<(NA * KP * NKP) / 256, 256, 0, stream>>>(dbuf, cutb, eh);

  float* vin = va; float* vout = vb;
  for (int l = 0; l < LL; l++) {
    k_gemm_qkv<<<NA / 16, 256, 0, stream>>>(sh_, Wqkvh + (size_t)l * FD * NQKV, qkv);
    k_attn<<<NA, 128, 0, stream>>>(qkv, eh, Weh + (size_t)l * NKP * NEKV,
                                   bek + (size_t)l * FD, bev + (size_t)l * 384,
                                   idxs, cutb, vecn, vin, vout, s_, sh_, o_);
    float* tmp = vin; vin = vout; vout = tmp;
  }
  k_readout<<<NA, 128, 0, stream>>>(o_, lng, lnb, Wo1, Wo2, batch, out);
}
